// HyperNet_67439576482086
// MI455X (gfx1250) — matrix operand order repaired and checked by reading, not compiled, not run on hardware
//
#include <hip/hip_runtime.h>

typedef __bf16 bf16;
typedef __attribute__((ext_vector_type(16))) __bf16 v16bf;
typedef __attribute__((ext_vector_type(8)))  __bf16 v8bf;
typedef __attribute__((ext_vector_type(8)))  float  v8f;
typedef __attribute__((ext_vector_type(4)))  unsigned int v4u;
typedef __attribute__((ext_vector_type(8)))  int v8i;
typedef __attribute__((ext_vector_type(4)))  int v4i;

#if defined(__has_builtin)
#  if __has_builtin(__builtin_amdgcn_tensor_load_to_lds)
#    define HAVE_TDM 1
#  endif
#endif

#define BB   4
#define NPT  8192
#define VV   2048
#define CP   256
#define CV   256
#define CPOS 32
#define LATD 128
#define BN   (BB*NPT)
#define EPSV 1e-5f

// ---- stats region layout (float offsets inside ws[0..32KB)) ----
enum {
  OF_PE_SUM=0, OF_PE_SQ=32,
  OF_S1_SUM=64, OF_S1_SQ=320,
  OF_S1B_SUM=576, OF_S1B_SQ=832,
  OF_Z_SUM=1088, OF_Z_SQ=1344,
  OF_PMIN=1600, OF_PMAX=1612,
  ACC_FLOATS=1600,                 // only [0..1600) needs zeroing
  OF_PE_MU=1664, OF_PE_SC=1696, OF_PE_SH=1728,
  OF_MU1=1760, OF_SC1=2016, OF_SH1=2272,
  OF_MU1B=2528, OF_SC1B=2784, OF_SH1B=3040,
  OF_MUZ=3296, OF_SCZ=3552, OF_SHZ=3808,
  OF_LAT=4064
};

// ---- ws byte offsets ----
#define WS_STATS 0
#define WS_VOX   32768          // B*V float4         = 131072 B
#define WS_IDX   163840         // B*N int            = 131072 B
#define WS_Y0    294912         // BN*32 f32          = 4 MB
#define WS_PACK  4489216        // packed B mats      = 1114112 B
#define WS_Y1    5603328        // BN*256 f32         = 32 MB
#define WS_Y1B   39157760       // BN*256 f32         = 32 MB  (end ~72.7 MB)

// packed-B element offsets (in bf16 elements, relative to WS_PACK)
#define P_W1HI 0
#define P_W1LO 73728
#define P_B1HI 147456
#define P_B1LO 221184
#define P_W2HI 294912
#define P_W2LO 360448
#define P_B2HI 425984
#define P_B2LO 491520

__device__ __forceinline__ void bsplit(float x, bf16& h, bf16& l) {
  h = (bf16)x;
  l = (bf16)(x - (float)h);
}
__device__ __forceinline__ float reluf(float x) { return fmaxf(x, 0.f); }

// ---------------- K0: per-batch bbox ----------------
__global__ void k_bbox(const float* __restrict__ p, float* __restrict__ stats) {
  int b = blockIdx.x, t = threadIdx.x;
  const float* pb = p + (size_t)b * NPT * 3;
  float mn[3] = {1e30f, 1e30f, 1e30f}, mx[3] = {-1e30f, -1e30f, -1e30f};
  for (int i = t; i < NPT; i += 256) {
    #pragma unroll
    for (int d = 0; d < 3; ++d) {
      float v = pb[i * 3 + d];
      mn[d] = fminf(mn[d], v); mx[d] = fmaxf(mx[d], v);
    }
  }
  __shared__ float s[6 * 256];
  #pragma unroll
  for (int d = 0; d < 3; ++d) { s[d*256 + t] = mn[d]; s[(3+d)*256 + t] = mx[d]; }
  __syncthreads();
  for (int off = 128; off > 0; off >>= 1) {
    if (t < off) {
      #pragma unroll
      for (int d = 0; d < 3; ++d) {
        s[d*256 + t]     = fminf(s[d*256 + t],     s[d*256 + t + off]);
        s[(3+d)*256 + t] = fmaxf(s[(3+d)*256 + t], s[(3+d)*256 + t + off]);
      }
    }
    __syncthreads();
  }
  if (t < 3) {
    stats[OF_PMIN + b*3 + t] = s[t*256];
    stats[OF_PMAX + b*3 + t] = s[(3 + t)*256];
  }
}

// ---------------- K1: voxel centers + validity + ||v||^2 ----------------
__global__ void k_voxprep(const int* __restrict__ vidx, const float* __restrict__ minc,
                          const float* __restrict__ gsz, const float* __restrict__ stats,
                          float4* __restrict__ vox) {
  int b = blockIdx.x, t = threadIdx.x;
  float gs = gsz[0];
  float m0 = minc[b*3+0], m1 = minc[b*3+1], m2 = minc[b*3+2];
  float n0 = stats[OF_PMIN + b*3 + 0], n1 = stats[OF_PMIN + b*3 + 1], n2 = stats[OF_PMIN + b*3 + 2];
  float x0 = stats[OF_PMAX + b*3 + 0], x1 = stats[OF_PMAX + b*3 + 1], x2 = stats[OF_PMAX + b*3 + 2];
  __shared__ int cnt;
  if (t == 0) cnt = 0;
  __syncthreads();
  int lc = 0;
  for (int v = t; v < VV; v += 256) {
    const int* q = vidx + ((size_t)b * VV + v) * 3;
    float x = (float)q[0] * gs + m0 + gs * 0.5f;
    float y = (float)q[1] * gs + m1 + gs * 0.5f;
    float z = (float)q[2] * gs + m2 + gs * 0.5f;
    bool valid = (x >= n0) & (x <= x0) & (y >= n1) & (y <= x1) & (z >= n2) & (z <= x2);
    lc += valid ? 1 : 0;
  }
  atomicAdd(&cnt, lc);
  __syncthreads();
  bool anyv = (cnt > 0);
  for (int v = t; v < VV; v += 256) {
    const int* q = vidx + ((size_t)b * VV + v) * 3;
    float x = (float)q[0] * gs + m0 + gs * 0.5f;
    float y = (float)q[1] * gs + m1 + gs * 0.5f;
    float z = (float)q[2] * gs + m2 + gs * 0.5f;
    bool valid = (x >= n0) & (x <= x0) & (y >= n1) & (y <= x1) & (z >= n2) & (z <= x2);
    float c = x*x + y*y + z*z + ((valid || !anyv) ? 0.f : 1e30f);
    vox[(size_t)b * VV + v] = make_float4(x, y, z, c);
  }
}

// ---------------- K2: fused 1-NN scan + positional pre-activation ----------------
// Voxel cache staged memory->LDS with one TDM descriptor (32 KB contiguous copy).
__global__ void k_nn(const float* __restrict__ p, const float4* __restrict__ vox,
                     int* __restrict__ idx, float* __restrict__ y0,
                     const float* __restrict__ embW, const float* __restrict__ embB) {
  __shared__ float4 sv[VV];               // 32 KB, LDS-broadcast reads
  int b = blockIdx.x >> 5, chunk = blockIdx.x & 31, t = threadIdx.x;
#ifdef HAVE_TDM
  if (t < 32) {                            // wave 0 drives the Tensor Data Mover
    unsigned lds_off = (unsigned)(uintptr_t)(&sv[0]);   // low 32 bits = LDS byte offset
    unsigned long long ga = (unsigned long long)(uintptr_t)(vox + (size_t)b * VV);
    // D# group0: count=1 | lds_addr | global_addr[56:0] | type=2
    v4u g0;
    g0[0] = 1u;
    g0[1] = lds_off;
    g0[2] = (unsigned)(ga & 0xffffffffu);
    g0[3] = (unsigned)((ga >> 32) & 0x01ffffffu) | 0x80000000u;
    // D# group1: data_size=8B; tensor_dim0=4096; tensor_dim1=1; tile_dim0=4096; tile_dim1=1;
    //            tensor_dim0_stride=4096  (4096 * 8B = 32 KB copy)
    v8i g1;
    g1[0] = (int)(3u << 16);          // data_size=3 (8B); no multicast, no pad/iterate
    g1[1] = (int)(0x1000u << 16);     // tensor_dim0[15:0] in bits[63:48]
    g1[2] = (int)(1u << 16);          // tensor_dim0[31:16]=0 | tensor_dim1[15:0]=1
    g1[3] = (int)(0x1000u << 16);     // tensor_dim1[31:16]=0 | tile_dim0=4096
    g1[4] = 1;                        // tile_dim1=1, tile_dim2=0
    g1[5] = 0x1000;                   // tensor_dim0_stride=4096
    g1[6] = 0;
    g1[7] = 0;
    v4i gz = {0, 0, 0, 0};
#  if __clang_major__ >= 23
    v8i gz8 = {0, 0, 0, 0, 0, 0, 0, 0};
    __builtin_amdgcn_tensor_load_to_lds(g0, g1, gz, gz, gz8, 0);
#  else
    __builtin_amdgcn_tensor_load_to_lds(g0, g1, gz, gz, 0);
#  endif
    __builtin_amdgcn_s_wait_tensorcnt(0);
  }
  __syncthreads();
#else
  for (int v = t; v < VV; v += 256) sv[v] = vox[(size_t)b * VV + v];
  __syncthreads();
#endif
  int g = b * NPT + chunk * 256 + t;
  float px = p[g*3], py = p[g*3+1], pz = p[g*3+2];
  float nx = -2.f * px, ny = -2.f * py, nz = -2.f * pz;
  float best = 3.4e38f; int bi = 0;
  #pragma unroll 4
  for (int v = 0; v < VV; ++v) {
    float4 q = sv[v];
    float s = fmaf(nx, q.x, q.w);
    s = fmaf(ny, q.y, s);
    s = fmaf(nz, q.z, s);
    if (s < best) { best = s; bi = v; }
  }
  idx[g] = bi;
  float4 nb = sv[bi];
  float rx = px - nb.x, ry = py - nb.y, rz = pz - nb.z;
  float* yr = y0 + (size_t)g * CPOS;
  #pragma unroll
  for (int c = 0; c < CPOS; ++c) {
    float v = fmaf(rx, embW[c], embB[c]);
    v = fmaf(ry, embW[CPOS + c], v);
    v = fmaf(rz, embW[2 * CPOS + c], v);
    yr[c] = v;
  }
}

// ---------------- K3: pack W matrices into WMMA-fragment-linear bf16 hi/lo ----------------
// Layout per matrix: [ktile][ntile][lane(32)][16 bf16], fragment element i <-> K = kt*32+(lane/16)*8+i+(i>=8?8:0),
// N = nt*16 + lane%16  (B matrix 32x16 bf16 layout, ISA 7.12.2 / sparse-B analogue).
__global__ void k_packB(const float* __restrict__ w1, const float* __restrict__ b1,
                        const float* __restrict__ w2, const float* __restrict__ b2,
                        bf16* __restrict__ pack) {
  int tid = blockIdx.x * 256 + threadIdx.x;
  if (tid >= 17408) return;
  const float* src; bf16 *hi, *lo; int frag;
  if (tid < 4608)       { src = w1; hi = pack + P_W1HI; lo = pack + P_W1LO; frag = tid; }
  else if (tid < 9216)  { src = b1; hi = pack + P_B1HI; lo = pack + P_B1LO; frag = tid - 4608; }
  else if (tid < 13312) { src = w2; hi = pack + P_W2HI; lo = pack + P_W2LO; frag = tid - 9216; }
  else                  { src = b2; hi = pack + P_B2HI; lo = pack + P_B2LO; frag = tid - 13312; }
  int kt = frag >> 9, rem = frag & 511, nt = rem >> 5, lane = rem & 31;
  int k0 = kt * 32 + (lane >> 4) * 8;
  int nc = nt * 16 + (lane & 15);
  #pragma unroll
  for (int i = 0; i < 16; ++i) {
    float x = src[(size_t)(k0 + i + (i >= 8 ? 8 : 0)) * 256 + nc];
    bf16 h, l; bsplit(x, h, l);
    hi[(size_t)frag * 16 + i] = h;
    lo[(size_t)frag * 16 + i] = l;
  }
}

// ---------------- column sum/sumsq reducer ----------------
__global__ void k_stats_cols(const float* __restrict__ src, float* __restrict__ sum_o,
                             float* __restrict__ sq_o, int cols, int rows_per_block) {
  int t = threadIdx.x;
  int c = t % cols;
  int stride = 256 / cols;
  int r = blockIdx.x * rows_per_block + t / cols;
  int rend = blockIdx.x * rows_per_block + rows_per_block;
  float s = 0.f, q = 0.f;
  for (; r < rend; r += stride) {
    float v = src[(size_t)r * cols + c];
    s += v; q += v * v;
  }
  atomicAdd(&sum_o[c], s);
  atomicAdd(&sq_o[c], q);
}

// ---------------- finalize CBN params: mu, scale=rsig*g[cond], shift=be[cond] ----------------
__global__ void k_finalize(float* __restrict__ stats, int sumoff, int sqoff,
                           const float* __restrict__ g, const float* __restrict__ be,
                           const int* __restrict__ condp,
                           int muoff, int scoff, int shoff, int nch) {
  int t = threadIdx.x;
  if (t >= nch) return;
  int cond = condp[0];
  float inv = 1.f / (float)BN;
  float mu  = stats[sumoff + t] * inv;
  float var = stats[sqoff + t] * inv - mu * mu;
  float rs  = rsqrtf(var + EPSV);
  stats[muoff + t] = mu;
  stats[scoff + t] = rs * g[cond * nch + t];
  stats[shoff + t] = be[cond * nch + t];
}

// ---------------- latent softmax vector ----------------
__global__ void k_lat(float* __restrict__ stats, const float* __restrict__ code,
                      const float* __restrict__ W1, const float* __restrict__ b1,
                      const float* __restrict__ W2, const float* __restrict__ b2) {
  __shared__ float h[256];
  __shared__ float red[256];
  int t = threadIdx.x;
  float a = b1[t];
  for (int k = 0; k < LATD; ++k) a = fmaf(code[k], W1[k * 256 + t], a);
  h[t] = reluf(a);
  __syncthreads();
  float v = b2[t];
  for (int j = 0; j < 256; ++j) v = fmaf(h[j], W2[j * 256 + t], v);
  red[t] = v; __syncthreads();
  for (int off = 128; off > 0; off >>= 1) { if (t < off) red[t] = fmaxf(red[t], red[t + off]); __syncthreads(); }
  float m = red[0]; __syncthreads();
  float e = expf(v - m);
  red[t] = e; __syncthreads();
  for (int off = 128; off > 0; off >>= 1) { if (t < off) red[t] += red[t + off]; __syncthreads(); }
  stats[OF_LAT + t] = e / red[0];
}

// ---------------- GEMM1: xe[BN,288] @ {w1,b1}[288,256] via bf16x3 WMMA ----------------
__global__ __launch_bounds__(256) void k_gemm1(
    const int* __restrict__ idx, const float* __restrict__ vfeat,
    const float* __restrict__ y0, const float* __restrict__ stats,
    const bf16* __restrict__ pack,
    const float* __restrict__ w1b, const float* __restrict__ b1b,
    float* __restrict__ y1, float* __restrict__ y1b) {
  extern __shared__ __align__(32) char smem[];
  const int AS = 296;                      // padded row stride (bf16 elements)
  bf16* Ahi = (bf16*)smem;
  bf16* Alo = Ahi + 64 * AS;
  int t = threadIdx.x;
  int rowbase = blockIdx.x * 64;
  {   // stage A tile: gather voxel feats + CBN'd positional embedding, split hi/lo
    int r = t >> 2, q = t & 3;
    int g = rowbase + r;
    int b = g >> 13;
    int vi = idx[g];
    const float* vf  = vfeat + ((size_t)(b * VV + vi)) * CV;
    const float* y0r = y0 + (size_t)g * CPOS;
    for (int j = 0; j < 72; ++j) {
      int c = q * 72 + j;
      float x;
      if (c < CV) x = vf[c];
      else {
        int cc = c - CV;
        x = reluf((y0r[cc] - stats[OF_PE_MU + cc]) * stats[OF_PE_SC + cc] + stats[OF_PE_SH + cc]);
      }
      bf16 h, l; bsplit(x, h, l);
      Ahi[r * AS + c] = h;
      Alo[r * AS + c] = l;
    }
  }
  __syncthreads();
  int lane = t & 31, w = t >> 5;
  int mt = w & 3, which = w >> 2;
  const bf16* Bhi = pack + (which ? P_B1HI : P_W1HI);
  const bf16* Blo = pack + (which ? P_B1LO : P_W1LO);
  const float* bias = which ? b1b : w1b;
  float* out = which ? y1b : y1;
  int half = lane >> 4;
  int arow = mt * 16 + (lane & 15);
  v8f acc[16] = {};
  for (int kt = 0; kt < 9; ++kt) {
    if (kt + 1 < 9) {   // warm next K-tile's B stream (global_prefetch_b8)
      __builtin_prefetch((const void*)(Bhi + ((size_t)((kt + 1) * 16) * 32 + lane) * 16), 0, 0);
      __builtin_prefetch((const void*)(Blo + ((size_t)((kt + 1) * 16) * 32 + lane) * 16), 0, 0);
    }
    const bf16* ah = Ahi + arow * AS + kt * 32 + half * 8;
    const bf16* al = Alo + arow * AS + kt * 32 + half * 8;
    v8bf ah0 = *(const v8bf*)ah;
    v8bf ah1 = *(const v8bf*)(ah + 16);
    v8bf al0 = *(const v8bf*)al;
    v8bf al1 = *(const v8bf*)(al + 16);
    v16bf fa_hi, fa_lo;
    #pragma unroll
    for (int i = 0; i < 8; ++i) {
      fa_hi[i] = ah0[i]; fa_hi[8 + i] = ah1[i];
      fa_lo[i] = al0[i]; fa_lo[8 + i] = al1[i];
    }
    #pragma unroll
    for (int nt = 0; nt < 16; ++nt) {
      size_t fo = ((size_t)(kt * 16 + nt) * 32 + lane) * 16;
      v16bf bh = *(const v16bf*)(Bhi + fo);
      v16bf bl = *(const v16bf*)(Blo + fo);
      acc[nt] = __builtin_amdgcn_wmma_f32_16x16x32_bf16(false, fa_hi, false, bh, (short)0, acc[nt], false, false);
      acc[nt] = __builtin_amdgcn_wmma_f32_16x16x32_bf16(false, fa_hi, false, bl, (short)0, acc[nt], false, false);
      acc[nt] = __builtin_amdgcn_wmma_f32_16x16x32_bf16(false, fa_lo, false, bh, (short)0, acc[nt], false, false);
    }
  }
  #pragma unroll
  for (int nt = 0; nt < 16; ++nt) {
    int c = nt * 16 + (lane & 15);
    float bi = bias[c];
    #pragma unroll
    for (int j = 0; j < 8; ++j) {
      int rr = mt * 16 + half * 8 + j;       // C/D: VGPR j -> M=j (+8 for lanes 16..31)
      out[(size_t)(rowbase + rr) * CP + c] = acc[nt][j] + bi;
    }
  }
}

// ---------------- GEMM2: relu(cbn(y1))@w2, relu(cbn(y1b))@b2  (in-place) ----------------
__global__ __launch_bounds__(256) void k_gemm2(
    const float* __restrict__ stats, const bf16* __restrict__ pack,
    const float* __restrict__ w2b, const float* __restrict__ b2b,
    float* __restrict__ y1, float* __restrict__ y1b) {
  extern __shared__ __align__(32) char smem[];
  const int AS = 264;
  bf16* A1hi = (bf16*)smem;
  bf16* A1lo = A1hi + 64 * AS;
  bf16* A2hi = A1lo + 64 * AS;
  bf16* A2lo = A2hi + 64 * AS;
  int t = threadIdx.x;
  int rowbase = blockIdx.x * 64;
  {
    int r = t >> 2, q = t & 3;
    int g = rowbase + r;
    const float* r1 = y1  + (size_t)g * CP;
    const float* r2 = y1b + (size_t)g * CP;
    for (int j = 0; j < 64; ++j) {
      int c = q * 64 + j;
      float x1 = reluf((r1[c] - stats[OF_MU1  + c]) * stats[OF_SC1  + c] + stats[OF_SH1  + c]);
      float x2 = reluf((r2[c] - stats[OF_MU1B + c]) * stats[OF_SC1B + c] + stats[OF_SH1B + c]);
      bf16 h, l;
      bsplit(x1, h, l); A1hi[r * AS + c] = h; A1lo[r * AS + c] = l;
      bsplit(x2, h, l); A2hi[r * AS + c] = h; A2lo[r * AS + c] = l;
    }
  }
  __syncthreads();
  int lane = t & 31, w = t >> 5;
  int mt = w & 3, which = w >> 2;
  bf16* Ah = which ? A2hi : A1hi;
  bf16* Al = which ? A2lo : A1lo;
  const bf16* Bhi = pack + (which ? P_B2HI : P_W2HI);
  const bf16* Blo = pack + (which ? P_B2LO : P_W2LO);
  const float* bias = which ? b2b : w2b;
  float* out = which ? y1b : y1;
  int half = lane >> 4;
  int arow = mt * 16 + (lane & 15);
  v8f acc[16] = {};
  for (int kt = 0; kt < 8; ++kt) {
    if (kt + 1 < 8) {
      __builtin_prefetch((const void*)(Bhi + ((size_t)((kt + 1) * 16) * 32 + lane) * 16), 0, 0);
      __builtin_prefetch((const void*)(Blo + ((size_t)((kt + 1) * 16) * 32 + lane) * 16), 0, 0);
    }
    const bf16* ah = Ah + arow * AS + kt * 32 + half * 8;
    const bf16* al = Al + arow * AS + kt * 32 + half * 8;
    v8bf ah0 = *(const v8bf*)ah;
    v8bf ah1 = *(const v8bf*)(ah + 16);
    v8bf al0 = *(const v8bf*)al;
    v8bf al1 = *(const v8bf*)(al + 16);
    v16bf fa_hi, fa_lo;
    #pragma unroll
    for (int i = 0; i < 8; ++i) {
      fa_hi[i] = ah0[i]; fa_hi[8 + i] = ah1[i];
      fa_lo[i] = al0[i]; fa_lo[8 + i] = al1[i];
    }
    #pragma unroll
    for (int nt = 0; nt < 16; ++nt) {
      size_t fo = ((size_t)(kt * 16 + nt) * 32 + lane) * 16;
      v16bf bh = *(const v16bf*)(Bhi + fo);
      v16bf bl = *(const v16bf*)(Blo + fo);
      acc[nt] = __builtin_amdgcn_wmma_f32_16x16x32_bf16(false, fa_hi, false, bh, (short)0, acc[nt], false, false);
      acc[nt] = __builtin_amdgcn_wmma_f32_16x16x32_bf16(false, fa_hi, false, bl, (short)0, acc[nt], false, false);
      acc[nt] = __builtin_amdgcn_wmma_f32_16x16x32_bf16(false, fa_lo, false, bh, (short)0, acc[nt], false, false);
    }
  }
  #pragma unroll
  for (int nt = 0; nt < 16; ++nt) {
    int c = nt * 16 + (lane & 15);
    float bi = bias[c];
    #pragma unroll
    for (int j = 0; j < 8; ++j) {
      int rr = mt * 16 + half * 8 + j;
      out[(size_t)(rowbase + rr) * CP + c] = acc[nt][j] + bi;
    }
  }
}

// ---------------- fused z = (y2*f + y2b)*lat, + stats ----------------
__global__ void k_zstats(float* __restrict__ y1, const float* __restrict__ y1b,
                         const float* __restrict__ f, float* __restrict__ stats) {
  int c = threadIdx.x;
  int r0 = blockIdx.x * 256;
  float latc = stats[OF_LAT + c];
  float s = 0.f, q = 0.f;
  for (int r = r0; r < r0 + 256; ++r) {
    size_t i = (size_t)r * CP + c;
    float z = (y1[i] * f[i] + y1b[i]) * latc;
    y1[i] = z;
    s += z; q += z * z;
  }
  atomicAdd(&stats[OF_Z_SUM + c], s);
  atomicAdd(&stats[OF_Z_SQ + c], q);
}

// ---------------- final: out = relu(cbn(z)) ----------------
__global__ void k_out(const float* __restrict__ z, const float* __restrict__ stats,
                      float* __restrict__ out) {
  int i = blockIdx.x * 256 + threadIdx.x;      // index over float4s
  float4 v = ((const float4*)z)[i];
  int c = (i & 63) * 4;
  float4 o;
  o.x = reluf((v.x - stats[OF_MUZ + c + 0]) * stats[OF_SCZ + c + 0] + stats[OF_SHZ + c + 0]);
  o.y = reluf((v.y - stats[OF_MUZ + c + 1]) * stats[OF_SCZ + c + 1] + stats[OF_SHZ + c + 1]);
  o.z = reluf((v.z - stats[OF_MUZ + c + 2]) * stats[OF_SCZ + c + 2] + stats[OF_SHZ + c + 2]);
  o.w = reluf((v.w - stats[OF_MUZ + c + 3]) * stats[OF_SCZ + c + 3] + stats[OF_SHZ + c + 3]);
  ((float4*)out)[i] = o;
}

extern "C" void kernel_launch(void* const* d_in, const int* in_sizes, int n_in,
                              void* d_out, int out_size, void* d_ws, size_t ws_size,
                              hipStream_t stream) {
  (void)in_sizes; (void)n_in; (void)out_size; (void)ws_size;
  const float* p           = (const float*)d_in[0];
  const float* f           = (const float*)d_in[1];
  const int*   voxel_idx   = (const int*)d_in[2];
  const float* voxel_feats = (const float*)d_in[3];
  const float* min_coords  = (const float*)d_in[4];
  const float* grid_size   = (const float*)d_in[5];
  const int*   condition   = (const int*)d_in[6];
  const float* emb_W = (const float*)d_in[7];  const float* emb_b = (const float*)d_in[8];
  const float* emb_g = (const float*)d_in[9];  const float* emb_be = (const float*)d_in[10];
  const float* w1_W = (const float*)d_in[11];  const float* w1_b = (const float*)d_in[12];
  const float* w_g  = (const float*)d_in[13];  const float* w_be = (const float*)d_in[14];
  const float* w2_W = (const float*)d_in[15];  const float* w2_b = (const float*)d_in[16];
  const float* b1_W = (const float*)d_in[17];  const float* b1_b = (const float*)d_in[18];
  const float* b_g  = (const float*)d_in[19];  const float* b_be = (const float*)d_in[20];
  const float* b2_W = (const float*)d_in[21];  const float* b2_b = (const float*)d_in[22];
  const float* lat_code = (const float*)d_in[23];
  const float* lat1_W = (const float*)d_in[24]; const float* lat1_b = (const float*)d_in[25];
  const float* lat2_W = (const float*)d_in[26]; const float* lat2_b = (const float*)d_in[27];
  const float* pt_g = (const float*)d_in[28];   const float* pt_be = (const float*)d_in[29];

  char* ws = (char*)d_ws;
  float*  stats = (float*) (ws + WS_STATS);
  float4* vox   = (float4*)(ws + WS_VOX);
  int*    idx   = (int*)   (ws + WS_IDX);
  float*  y0    = (float*) (ws + WS_Y0);
  bf16*   pack  = (bf16*)  (ws + WS_PACK);
  float*  y1    = (float*) (ws + WS_Y1);
  float*  y1b   = (float*) (ws + WS_Y1B);

  hipMemsetAsync(stats, 0, ACC_FLOATS * sizeof(float), stream);
  k_bbox<<<BB, 256, 0, stream>>>(p, stats);
  k_voxprep<<<BB, 256, 0, stream>>>(voxel_idx, min_coords, grid_size, stats, vox);
  k_nn<<<BB * 32, 256, 0, stream>>>(p, vox, idx, y0, emb_W, emb_b);
  k_packB<<<68, 256, 0, stream>>>(w1_W, b1_W, w2_W, b2_W, pack);
  k_stats_cols<<<64, 256, 0, stream>>>(y0, stats + OF_PE_SUM, stats + OF_PE_SQ, CPOS, BN / 64);
  k_lat<<<1, 256, 0, stream>>>(stats, lat_code, lat1_W, lat1_b, lat2_W, lat2_b);
  k_finalize<<<1, 32, 0, stream>>>(stats, OF_PE_SUM, OF_PE_SQ, emb_g, emb_be, condition,
                                   OF_PE_MU, OF_PE_SC, OF_PE_SH, CPOS);
  k_gemm1<<<BN / 64, 256, 2 * 64 * 296 * (int)sizeof(bf16), stream>>>(
      idx, voxel_feats, y0, stats, pack, w1_b, b1_b, y1, y1b);
  k_stats_cols<<<128, 256, 0, stream>>>(y1,  stats + OF_S1_SUM,  stats + OF_S1_SQ,  CP, BN / 128);
  k_stats_cols<<<128, 256, 0, stream>>>(y1b, stats + OF_S1B_SUM, stats + OF_S1B_SQ, CP, BN / 128);
  k_finalize<<<1, 256, 0, stream>>>(stats, OF_S1_SUM, OF_S1_SQ, w_g, w_be, condition,
                                    OF_MU1, OF_SC1, OF_SH1, CP);
  k_finalize<<<1, 256, 0, stream>>>(stats, OF_S1B_SUM, OF_S1B_SQ, b_g, b_be, condition,
                                    OF_MU1B, OF_SC1B, OF_SH1B, CP);
  k_gemm2<<<BN / 64, 256, 4 * 64 * 264 * (int)sizeof(bf16), stream>>>(
      stats, pack, w2_b, b2_b, y1, y1b);
  k_zstats<<<128, 256, 0, stream>>>(y1, y1b, f, stats);
  k_finalize<<<1, 256, 0, stream>>>(stats, OF_Z_SUM, OF_Z_SQ, pt_g, pt_be, condition,
                                    OF_MUZ, OF_SCZ, OF_SHZ, CP);
  k_out<<<BN * CP / 1024, 256, 0, stream>>>(y1, stats, (float*)d_out);
}
